// GATConv_9689446220155
// MI455X (gfx1250) — matrix operand order repaired and checked by reading, not compiled, not run on hardware
//
#include <hip/hip_runtime.h>
#include <hip/hip_bf16.h>

#define N_SRC 100000
#define N_TGT 100000
#define IN_F  256
#define HID   128
#define NEDGE 1600000

typedef __attribute__((ext_vector_type(16))) __bf16 v16bf;
typedef __attribute__((ext_vector_type(8)))  float  v8f;

union FragBF {
    uint4 q[2];   // 2 x 16B = 16 bf16
    v16bf v;
};

__device__ __forceinline__ unsigned short f32_to_bf16_rne(float f) {
    unsigned int u = __float_as_uint(f);
    u += 0x7FFFu + ((u >> 16) & 1u);     // round-to-nearest-even
    return (unsigned short)(u >> 16);
}

// Order-preserving float -> uint key (monotonic), for atomicMax-based segment max.
__device__ __forceinline__ unsigned int fkey(float x) {
    unsigned int u = __float_as_uint(x);
    return u ^ ((((int)u) >> 31) | 0x80000000u);
}
__device__ __forceinline__ float unkey(unsigned int k) {
    unsigned int u = (k & 0x80000000u) ? (k ^ 0x80000000u) : ~k;
    return __uint_as_float(u);
}

// ---------------- init kernels ----------------
__global__ void conv_w_kernel(const float* __restrict__ W, unsigned short* __restrict__ Wbf, int n) {
    int i = blockIdx.x * blockDim.x + threadIdx.x;
    if (i < n) Wbf[i] = f32_to_bf16_rne(W[i]);
}

__global__ void init_out_kernel(float* __restrict__ out, const float* __restrict__ bias, int n) {
    int i = blockIdx.x * blockDim.x + threadIdx.x;
    if (i < n) out[i] = bias[i & (HID - 1)];
}

__global__ void init_seg_kernel(unsigned int* __restrict__ mkey, float* __restrict__ denom, int n) {
    int i = blockIdx.x * blockDim.x + threadIdx.x;
    if (i < n) { mkey[i] = 0u; denom[i] = 0.0f; }
}

// ---------------- WMMA GEMM: HL = X @ W^T + b_lin ----------------
// grid.x = M/16 workgroups, 256 threads (8 waves). Wave w computes N-tile w (16 cols).
// A tile (16 x 256) staged in LDS as bf16, shared by all 8 waves.
__global__ __launch_bounds__(256) void gemm_hl_kernel(const float* __restrict__ X,
                                                      const unsigned short* __restrict__ Wbf,
                                                      const float* __restrict__ b_lin,
                                                      float* __restrict__ HL) {
    __shared__ __align__(16) unsigned short Abf[16 * IN_F];   // 8 KB

    const int m0 = blockIdx.x * 16;
    const int t  = threadIdx.x;

    // Stage + convert A tile: thread t -> row t/16, 16-col segment t%16.
    {
        const int row = t >> 4, seg = t & 15;
        const float* src = X + (size_t)(m0 + row) * IN_F + seg * 16;
        unsigned short* dst = Abf + row * IN_F + seg * 16;
#pragma unroll
        for (int i = 0; i < 16; ++i) dst[i] = f32_to_bf16_rne(src[i]);
    }
    __syncthreads();

    const int wave   = t >> 5;        // 0..7  -> N tile
    const int lane   = t & 31;
    const int nlo    = lane & 15;
    const int hiHalf = lane >> 4;     // 0 or 1
    const int col    = wave * 16 + nlo;   // global HID column (0..127)

    v8f c = {};
#pragma unroll
    for (int kt = 0; kt < IN_F / 32; ++kt) {
        FragBF a, b;
        // A frag (16x32 bf16): lane=row nlo; halves 0..7 = K[kb..kb+7], 8..15 = K[kb+16..kb+23]
        const int akb = kt * 32 + hiHalf * 8;
        const unsigned short* ap = Abf + nlo * IN_F + akb;
        a.q[0] = *(const uint4*)(ap);
        a.q[1] = *(const uint4*)(ap + 16);
        // B frag (32x16 bf16): lane=col; halves 0..7 = K[kb..kb+7], 8..15 = K[kb+16..kb+23]
        const int bkb = kt * 32 + hiHalf * 8;
        const unsigned short* bp = Wbf + (size_t)col * IN_F + bkb;
        b.q[0] = *(const uint4*)(bp);
        b.q[1] = *(const uint4*)(bp + 16);

        c = __builtin_amdgcn_wmma_f32_16x16x32_bf16(false, a.v, false, b.v,
                                                    (short)0, c, false, false);
    }

    // C/D layout: v[i] = C[mbase+i][nlo], mbase = 8*hiHalf
    const float bl = b_lin[col];
    const int   mb = hiHalf * 8;
#pragma unroll
    for (int i = 0; i < 8; ++i)
        HL[(size_t)(m0 + mb + i) * HID + col] = c[i] + bl;
}

// ---------------- score = HL @ w (per-row dot over HID) ----------------
__global__ __launch_bounds__(256) void score_kernel(const float* __restrict__ HLbuf,
                                                    const float* __restrict__ wvec,
                                                    float* __restrict__ out, int n) {
    const int g    = blockIdx.x * blockDim.x + threadIdx.x;
    const int row  = g >> 5;
    const int lane = threadIdx.x & 31;
    if (row >= n) return;
    const float4 h = *(const float4*)(HLbuf + (size_t)row * HID + lane * 4);
    const float4 w = *(const float4*)(wvec + lane * 4);
    float s = h.x * w.x + h.y * w.y + h.z * w.z + h.w * w.w;
#pragma unroll
    for (int off = 16; off > 0; off >>= 1) s += __shfl_xor(s, off, 32);
    if (lane == 0) out[row] = s;
}

// ---------------- edge: e = tanh(s[si] + t[ti] + b), segment max ----------------
__global__ void edge_tanh_max_kernel(const float* __restrict__ s_score,
                                     const float* __restrict__ t_score,
                                     const int* __restrict__ edges,
                                     const float* __restrict__ att_b,
                                     float* __restrict__ e_buf,
                                     unsigned int* __restrict__ mkey, int nE) {
    const int i = blockIdx.x * blockDim.x + threadIdx.x;
    if (i >= nE) return;
    const int si = edges[i];
    const int ti = edges[nE + i];
    const float e = tanhf(s_score[si] + t_score[ti] + att_b[0]);
    e_buf[i] = e;
    atomicMax(mkey + si, fkey(e));
}

// ---------------- edge: e_exp = exp(e - m[si]), segment sum ----------------
__global__ void edge_exp_sum_kernel(float* __restrict__ e_buf,
                                    const unsigned int* __restrict__ mkey,
                                    const int* __restrict__ edges,
                                    float* __restrict__ denom, int nE) {
    const int i = blockIdx.x * blockDim.x + threadIdx.x;
    if (i >= nE) return;
    const int si = edges[i];
    const float m  = unkey(mkey[si]);
    const float ex = __expf(e_buf[i] - m);
    e_buf[i] = ex;
    atomicAdd(denom + si, ex);
}

// ---------------- heavy scatter: out[si] += (e_exp/denom[si]) * tgt_hl[ti] ----------------
// One wave per edge; lane handles 4 of 128 HID channels.
__global__ __launch_bounds__(256) void scatter_kernel(const float* __restrict__ e_exp,
                                                      const float* __restrict__ denom,
                                                      const int* __restrict__ edges,
                                                      const float* __restrict__ tgt_hl,
                                                      float* __restrict__ out, int nE) {
    const int g    = blockIdx.x * blockDim.x + threadIdx.x;
    const int edge = g >> 5;
    const int lane = threadIdx.x & 31;
    if (edge >= nE) return;
    const int si = edges[edge];
    const int ti = edges[nE + edge];
    const float att = e_exp[edge] / denom[si];
    const float4 h = *(const float4*)(tgt_hl + (size_t)ti * HID + lane * 4);
    float* o = out + (size_t)si * HID + lane * 4;
    atomicAdd(o + 0, att * h.x);
    atomicAdd(o + 1, att * h.y);
    atomicAdd(o + 2, att * h.z);
    atomicAdd(o + 3, att * h.w);
}

extern "C" void kernel_launch(void* const* d_in, const int* in_sizes, int n_in,
                              void* d_out, int out_size, void* d_ws, size_t ws_size,
                              hipStream_t stream) {
    const float* source_h = (const float*)d_in[0];
    const float* target_h = (const float*)d_in[1];
    const int*   edges    = (const int*)d_in[2];
    const float* W        = (const float*)d_in[3];
    const float* b_lin    = (const float*)d_in[4];
    const float* att_w    = (const float*)d_in[5];
    const float* att_b    = (const float*)d_in[6];
    const float* bias     = (const float*)d_in[7];
    float* out = (float*)d_out;

    // Workspace layout (all offsets 16B-aligned)
    float* ws      = (float*)d_ws;
    float* src_hl  = ws;                                   // N_SRC*HID
    float* tgt_hl  = src_hl + (size_t)N_SRC * HID;         // N_TGT*HID
    float* s_score = tgt_hl + (size_t)N_TGT * HID;         // N_SRC
    float* t_score = s_score + N_SRC;                      // N_TGT
    float* e_buf   = t_score + N_TGT;                      // NEDGE (e, then e_exp in place)
    float* denom   = e_buf + NEDGE;                        // N_SRC
    unsigned int*   mkey = (unsigned int*)(denom + N_SRC); // N_SRC
    unsigned short* Wbf  = (unsigned short*)(mkey + N_SRC);// HID*IN_F bf16

    // init / convert
    conv_w_kernel<<<(HID * IN_F + 255) / 256, 256, 0, stream>>>(W, Wbf, HID * IN_F);
    init_out_kernel<<<((size_t)N_SRC * HID + 255) / 256, 256, 0, stream>>>(out, bias, N_SRC * HID);
    init_seg_kernel<<<(N_SRC + 255) / 256, 256, 0, stream>>>(mkey, denom, N_SRC);

    // WMMA GEMMs: hl = X @ W^T + b_lin
    gemm_hl_kernel<<<N_SRC / 16, 256, 0, stream>>>(source_h, Wbf, b_lin, src_hl);
    gemm_hl_kernel<<<N_TGT / 16, 256, 0, stream>>>(target_h, Wbf, b_lin, tgt_hl);

    // per-row attention scores
    score_kernel<<<(N_SRC * 32 + 255) / 256, 256, 0, stream>>>(src_hl, att_w,       s_score, N_SRC);
    score_kernel<<<(N_TGT * 32 + 255) / 256, 256, 0, stream>>>(tgt_hl, att_w + HID, t_score, N_TGT);

    // edge softmax pipeline
    edge_tanh_max_kernel<<<(NEDGE + 255) / 256, 256, 0, stream>>>(s_score, t_score, edges, att_b, e_buf, mkey, NEDGE);
    edge_exp_sum_kernel<<<(NEDGE + 255) / 256, 256, 0, stream>>>(e_buf, mkey, edges, denom, NEDGE);

    // weighted scatter-aggregate (one wave per edge)
    scatter_kernel<<<((size_t)NEDGE * 32 + 255) / 256, 256, 0, stream>>>(e_buf, denom, edges, tgt_hl, out, NEDGE);
}
